// Encoder_3770981286132
// MI455X (gfx1250) — matrix operand order repaired and checked by reading, not compiled, not run on hardware
//
#include <hip/hip_runtime.h>
#include <cmath>

typedef __attribute__((ext_vector_type(16))) _Float16 v16h;
typedef __attribute__((ext_vector_type(8)))  float    v8f;

#define Bv   128
#define Tv   512
#define Dv   256
#define Hv   256
#define DEP  3
#define NINC 2

__device__ inline v8f zero8() {
  v8f z;
#pragma unroll
  for (int i = 0; i < 8; ++i) z[i] = 0.f;
  return z;
}

__device__ inline v8f wmma16(v16h a, v16h b, v8f c) {
  return __builtin_amdgcn_wmma_f32_16x16x32_f16(false, a, false, b, (short)0, c,
                                                false, false);
}

__device__ inline float sigf(float x) { return 1.f / (1.f + __expf(-x)); }

// Build an A fragment (16x32 f16) from two contiguous 8-float runs, masked.
// Lane layout (ISA): rows = lane&15; K = kgA + {0..7} and kgA + {16..23}.
__device__ inline v16h load_a_frag(const float* __restrict__ ab, float m) {
  float4 u0 = ((const float4*)ab)[0];
  float4 u1 = ((const float4*)ab)[1];
  float4 u2 = ((const float4*)(ab + 16))[0];
  float4 u3 = ((const float4*)(ab + 16))[1];
  v16h a;
  a[0] = (_Float16)(u0.x * m);  a[1] = (_Float16)(u0.y * m);
  a[2] = (_Float16)(u0.z * m);  a[3] = (_Float16)(u0.w * m);
  a[4] = (_Float16)(u1.x * m);  a[5] = (_Float16)(u1.y * m);
  a[6] = (_Float16)(u1.z * m);  a[7] = (_Float16)(u1.w * m);
  a[8] = (_Float16)(u2.x * m);  a[9] = (_Float16)(u2.y * m);
  a[10] = (_Float16)(u2.z * m); a[11] = (_Float16)(u2.w * m);
  a[12] = (_Float16)(u3.x * m); a[13] = (_Float16)(u3.y * m);
  a[14] = (_Float16)(u3.z * m); a[15] = (_Float16)(u3.w * m);
  return a;
}

// ---------------------------------------------------------------- embedding
__global__ void embed_kernel(const int* __restrict__ tokens,
                             const float* __restrict__ E,
                             float* __restrict__ x) {
  int i = blockIdx.x * 256 + threadIdx.x;
  int row = i >> 8;
  int d = i & 255;
  x[i] = E[tokens[row] * Dv + d];
}

__global__ void zero_kernel(float* __restrict__ p, int n) {
  int i = blockIdx.x * 256 + threadIdx.x;
  if (i < n) p[i] = 0.f;
}

// -------------------------------------------- conv weight -> packed fragments
// out layout: [blk][tap][kc8][nt][lane][e], 512 halves per fragment block.
__global__ void pack_convw_kernel(const float* __restrict__ W,
                                  _Float16* __restrict__ out, int KS) {
  int i = blockIdx.x * 256 + threadIdx.x;
  if (i >= NINC * KS * 8 * 16 * 512) return;
  int e = i & 15;
  int lane = (i >> 4) & 31;
  int nt = (i >> 9) & 15;
  int kc8 = (i >> 13) & 7;
  int rest = i >> 16;        // blk*KS + tap
  int kgB = (lane >> 4) << 3;
  int k = kc8 * 32 + kgB + (e & 7) + ((e >> 3) << 4);
  int n = nt * 16 + (lane & 15);
  out[i] = (_Float16)W[((size_t)rest * Dv + k) * Dv + n];
}

// -------------------------------------------- LSTM weights -> packed fragments
// out layout: [dir*DEP+layer][kc16][g][nt][lane][e]; K 0..255 = Wx, 256..511 = Wh.
__global__ void pack_lstmw_kernel(const float* __restrict__ Wx,
                                  const float* __restrict__ Wh,
                                  _Float16* __restrict__ out) {
  int i = blockIdx.x * 256 + threadIdx.x;
  int e = i & 15;
  int lane = (i >> 4) & 31;
  int nt = (i >> 9) & 15;
  int g = (i >> 13) & 3;
  int kc16 = (i >> 15) & 15;
  int dl = i >> 19;
  if (dl >= 2 * DEP) return;
  int kgB = (lane >> 4) << 3;
  int k = kc16 * 32 + kgB + (e & 7) + ((e >> 3) << 4);
  int col = g * Hv + nt * 16 + (lane & 15);
  float v = (k < Dv) ? Wx[((size_t)dl * Dv + k) * 4 * Hv + col]
                     : Wh[((size_t)dl * Hv + (k - Dv)) * 4 * Hv + col];
  out[i] = (_Float16)v;
}

// ------------------------------------------------- one conv branch (templated)
template <int KS>
__device__ inline v8f conv_branch(const float* __restrict__ x, int b, int t0,
                                  int arow, int kgA,
                                  const _Float16* __restrict__ wp, int lane,
                                  int nt) {
  v8f acc = zero8();
  for (int tap = 0; tap < KS; ++tap) {
    int trow = t0 + arow + tap - (KS >> 1);
    bool inb = (trow >= 0) && (trow < Tv);
    float m = inb ? 1.f : 0.f;
    const float* xrow = x + ((size_t)b * Tv + (inb ? trow : 0)) * Dv;
    for (int kc8 = 0; kc8 < 8; ++kc8) {
      v16h a = load_a_frag(xrow + kc8 * 32 + kgA, m);
      const v16h* bp =
          (const v16h*)(wp + (((size_t)tap * 8 + kc8) * 16 + nt) * 512);
      acc = wmma16(a, bp[lane], acc);
    }
  }
  return acc;
}

// ------------------------------------------------- inception conv block (WMMA)
__global__ void __launch_bounds__(128)
conv_block_kernel(const float* __restrict__ x,
                  const _Float16* __restrict__ w3p,
                  const _Float16* __restrict__ w5p,
                  const _Float16* __restrict__ w7p,
                  const float* __restrict__ bconv,
                  float* __restrict__ feats, int blk) {
  int wave = blockIdx.x * (blockDim.x >> 5) + (threadIdx.x >> 5);
  int lane = threadIdx.x & 31;
  int mt = wave >> 4;
  int nt = wave & 15;
  int b = mt >> 5;
  int t0 = (mt & 31) << 4;
  int arow = lane & 15;
  int kgA = (lane >> 4) << 3;
  int ncol = nt * 16 + (lane & 15);

  v8f a3 = conv_branch<3>(x, b, t0, arow, kgA,
                          w3p + (size_t)blk * 3 * 8 * 16 * 512, lane, nt);
  v8f a5 = conv_branch<5>(x, b, t0, arow, kgA,
                          w5p + (size_t)blk * 5 * 8 * 16 * 512, lane, nt);
  v8f a7 = conv_branch<7>(x, b, t0, arow, kgA,
                          w7p + (size_t)blk * 7 * 8 * 16 * 512, lane, nt);

  float b0 = bconv[((size_t)blk * 3 + 0) * Dv + ncol];
  float b1 = bconv[((size_t)blk * 3 + 1) * Dv + ncol];
  float b2 = bconv[((size_t)blk * 3 + 2) * Dv + ncol];
#pragma unroll
  for (int r = 0; r < 8; ++r) {
    int M = r + ((lane >> 4) << 3);
    size_t grow = (size_t)mt * 16 + M;
    float v = x[grow * Dv + ncol] + tanhf(a3[r] + b0) + tanhf(a5[r] + b1) +
              tanhf(a7[r] + b2);
    feats[grow * Dv + ncol] = v;
  }
}

// ------------------------------------------------------------------ layernorm
__global__ void layernorm_kernel(const float* __restrict__ feats,
                                 const float* __restrict__ gamma,
                                 const float* __restrict__ beta,
                                 float* __restrict__ xout, int blk) {
  __shared__ float red[256];
  size_t row = blockIdx.x;
  int d = threadIdx.x;
  float v = feats[row * Dv + d];
  red[d] = v;
  __syncthreads();
  for (int s = 128; s > 0; s >>= 1) {
    if (d < s) red[d] += red[d + s];
    __syncthreads();
  }
  float mu = red[0] * (1.f / Dv);
  __syncthreads();
  float dv = v - mu;
  red[d] = dv * dv;
  __syncthreads();
  for (int s = 128; s > 0; s >>= 1) {
    if (d < s) red[d] += red[d + s];
    __syncthreads();
  }
  float rstd = rsqrtf(red[0] * (1.f / Dv) + 1e-3f);
  xout[row * Dv + d] = dv * rstd * gamma[blk * Dv + d] + beta[blk * Dv + d];
}

// ------------------------------------------------------- fused LSTM cell step
// hbuf: [parity][dir][DEP][B][H], cbuf: [dir][DEP][B][H], curbuf: [ping][dir][B][D]
__global__ void __launch_bounds__(128)
lstm_step_kernel(const float* __restrict__ x,
                 const _Float16* __restrict__ wpacked,
                 const float* __restrict__ blstm,
                 float* __restrict__ hbuf, float* __restrict__ cbuf,
                 float* __restrict__ curbuf, float* __restrict__ out, int s,
                 int layer) {
  int wave = blockIdx.x * (blockDim.x >> 5) + (threadIdx.x >> 5);
  int lane = threadIdx.x & 31;
  int dir = wave >> 7;
  int rem = wave & 127;
  int mt = rem >> 4;
  int nt = rem & 15;
  int tpos = dir ? (Tv - 1 - s) : s;
  int arow = lane & 15;
  int kgA = (lane >> 4) << 3;
  int ncol = nt * 16 + (lane & 15);
  int brow = mt * 16 + arow;

  int pin = (layer + 1) & 1;
  int pout = layer & 1;
  int hpin = (s & 1) ^ 1;
  int hpout = s & 1;
  int dl = dir * DEP + layer;

  const float* curin = (layer == 0)
      ? (x + ((size_t)brow * Tv + tpos) * Dv)
      : (curbuf + ((size_t)(pin * 2 + dir) * Bv + brow) * Dv);
  const float* hin =
      hbuf + ((size_t)((hpin * 2 + dir) * DEP + layer) * Bv + brow) * Hv;
  const _Float16* wl = wpacked + (size_t)dl * 16 * 4 * 16 * 512;

  v8f acc0 = zero8(), acc1 = zero8(), acc2 = zero8(), acc3 = zero8();
  for (int kc16 = 0; kc16 < 16; ++kc16) {
    const float* abase =
        (kc16 < 8) ? (curin + kc16 * 32) : (hin + kc16 * 32 - Dv);
    v16h a = load_a_frag(abase + kgA, 1.f);
    const _Float16* wk = wl + (((size_t)kc16 * 4) * 16 + nt) * 512 + lane * 16;
    acc0 = wmma16(a, *(const v16h*)(wk + 0 * 16 * 512), acc0);
    acc1 = wmma16(a, *(const v16h*)(wk + 1 * 16 * 512), acc1);
    acc2 = wmma16(a, *(const v16h*)(wk + 2 * 16 * 512), acc2);
    acc3 = wmma16(a, *(const v16h*)(wk + 3 * 16 * 512), acc3);
  }

  float* cptr = cbuf + (size_t)dl * Bv * Hv;
  float* hout = hbuf + (size_t)((hpout * 2 + dir) * DEP + layer) * Bv * Hv;
  float* curout = curbuf + (size_t)(pout * 2 + dir) * Bv * Dv;
  const float* bvec = blstm + (size_t)dl * 4 * Hv;
  float bi = bvec[0 * Hv + ncol];
  float bf = bvec[1 * Hv + ncol];
  float bg = bvec[2 * Hv + ncol];
  float bo = bvec[3 * Hv + ncol];

#pragma unroll
  for (int r = 0; r < 8; ++r) {
    int M = r + ((lane >> 4) << 3);
    int b = mt * 16 + M;
    float zi = acc0[r] + bi;
    float zf = acc1[r] + bf;
    float zg = acc2[r] + bg;
    float zo = acc3[r] + bo;
    size_t idx = (size_t)b * Hv + ncol;
    float cnew = sigf(zf) * cptr[idx] + sigf(zi) * tanhf(zg);
    float hnew = sigf(zo) * tanhf(cnew);
    cptr[idx] = cnew;
    hout[idx] = hnew;
    float curold = (layer == 0)
        ? x[((size_t)b * Tv + tpos) * Dv + ncol]
        : curbuf[((size_t)(pin * 2 + dir) * Bv + b) * Dv + ncol];
    float curnew = curold + hnew;
    curout[(size_t)b * Dv + ncol] = curnew;
    if (layer == DEP - 1)
      out[((size_t)b * Tv + tpos) * (2 * Hv) + dir * Hv + ncol] = curnew;
  }
}

// ----------------------------------------------------------------------- host
extern "C" void kernel_launch(void* const* d_in, const int* in_sizes, int n_in,
                              void* d_out, int out_size, void* d_ws,
                              size_t ws_size, hipStream_t stream) {
  (void)in_sizes; (void)n_in; (void)out_size; (void)ws_size;
  const int* tokens = (const int*)d_in[0];
  const float* E = (const float*)d_in[1];
  const float* W3 = (const float*)d_in[2];
  const float* W5 = (const float*)d_in[3];
  const float* W7 = (const float*)d_in[4];
  const float* bconv = (const float*)d_in[5];
  const float* gamma = (const float*)d_in[6];
  const float* beta = (const float*)d_in[7];
  const float* Wx = (const float*)d_in[8];
  const float* Wh = (const float*)d_in[9];
  const float* blstm = (const float*)d_in[10];
  float* out = (float*)d_out;

  const size_t BT = (size_t)Bv * Tv;
  char* ws = (char*)d_ws;
  size_t o = 0;
  auto take = [&](size_t bytes) -> char* {
    char* p = ws + o;
    o += (bytes + 255) & ~(size_t)255;
    return p;
  };
  float* x      = (float*)take(BT * Dv * 4);
  float* feats  = (float*)take(BT * Dv * 4);
  _Float16* w3p = (_Float16*)take((size_t)NINC * 3 * Dv * Dv * 2);
  _Float16* w5p = (_Float16*)take((size_t)NINC * 5 * Dv * Dv * 2);
  _Float16* w7p = (_Float16*)take((size_t)NINC * 7 * Dv * Dv * 2);
  _Float16* wlp = (_Float16*)take((size_t)2 * DEP * 16 * 4 * 16 * 512 * 2);
  float* hbuf   = (float*)take((size_t)2 * 2 * DEP * Bv * Hv * 4);
  float* cbuf   = (float*)take((size_t)2 * DEP * Bv * Hv * 4);
  float* curbuf = (float*)take((size_t)2 * 2 * Bv * Dv * 4);

  // 1) embedding gather
  embed_kernel<<<(int)(BT * Dv / 256), 256, 0, stream>>>(tokens, E, x);

  // 2) pack weights into f16 WMMA fragment order
  pack_convw_kernel<<<(NINC * 3 * Dv * Dv + 255) / 256, 256, 0, stream>>>(W3, w3p, 3);
  pack_convw_kernel<<<(NINC * 5 * Dv * Dv + 255) / 256, 256, 0, stream>>>(W5, w5p, 5);
  pack_convw_kernel<<<(NINC * 7 * Dv * Dv + 255) / 256, 256, 0, stream>>>(W7, w7p, 7);
  pack_lstmw_kernel<<<(2 * DEP) << 11, 256, 0, stream>>>(Wx, Wh, wlp);

  // 3) zero h (both parities) + c states (contiguous allocations)
  {
    int n = 2 * 2 * DEP * Bv * Hv + 2 * DEP * Bv * Hv;
    zero_kernel<<<(n + 255) / 256, 256, 0, stream>>>(hbuf, n);
  }

  // 4) inception blocks
  for (int blk = 0; blk < NINC; ++blk) {
    conv_block_kernel<<<(int)(BT / 16 * 16 / 4), 128, 0, stream>>>(
        x, w3p, w5p, w7p, bconv, feats, blk);
    layernorm_kernel<<<(int)BT, 256, 0, stream>>>(feats, gamma, beta, x, blk);
  }

  // 5) bidirectional stacked LSTM
  for (int s = 0; s < Tv; ++s)
    for (int l = 0; l < DEP; ++l)
      lstm_step_kernel<<<64, 128, 0, stream>>>(x, wlp, blstm, hbuf, cbuf,
                                               curbuf, out, s, l);

  // 6) final states
  size_t seqFloats = BT * 2 * Hv;
  size_t stFloats = (size_t)2 * DEP * Bv * Hv;
  (void)hipMemcpyAsync(out + seqFloats, hbuf + stFloats, stFloats * 4,
                       hipMemcpyDeviceToDevice, stream);
  (void)hipMemcpyAsync(out + seqFloats + stFloats, cbuf, stFloats * 4,
                       hipMemcpyDeviceToDevice, stream);
}
